// V8Model_23579370455247
// MI455X (gfx1250) — matrix operand order repaired and checked by reading, not compiled, not run on hardware
//
#include <hip/hip_runtime.h>
#include <cmath>

// Problem constants (from the reference)
#define BS_    8
#define T_     2048
#define C_     8
#define DM_    256
#define DCC_   32
#define N_     512
#define MC_    32
#define AE_    32
#define NB_    64            // T_/AE_
#define HID_   128
#define OBSD_  132
#define OBSP_  160           // OBS padded to multiple of 32 (WMMA K)
#define ADIM_  290
#define ADIMP_ 304           // ADIM padded to multiple of 16 (WMMA N)
#define NTHR_  1024          // 32 waves of 32
#define MT_    128           // neuron tile for MLP GEMMs
#define TILE_  (C_ * DM_)    // 2048 elems / 8 KB per timestep tile
#define LOG2PI_F 1.8378770664093453f

typedef __attribute__((ext_vector_type(16))) _Float16 v16h;
typedef __attribute__((ext_vector_type(8)))  float    v8f;

// ---------------- optional CDNA5 fast paths (compile-probe guarded) ---------
#if defined(__AMDGCN__) && __has_builtin(__builtin_amdgcn_tensor_load_to_lds) && \
    __has_builtin(__builtin_amdgcn_tensor_store_from_lds) && \
    __has_builtin(__builtin_amdgcn_s_wait_tensorcnt)
#define USE_TDM 1
#else
#define USE_TDM 0
#endif

__device__ inline float fast_tanh(float x) {
#if __has_builtin(__builtin_amdgcn_tanhf)
  return __builtin_amdgcn_tanhf(x);
#elif __has_builtin(__builtin_amdgcn_tanh_f32)
  return __builtin_amdgcn_tanh_f32(x);
#else
  return tanhf(x);
#endif
}

// ---------------- LDS layout (dynamic shared, ~242 KB < 320 KB/WGP) ---------
struct __align__(16) Smem {
  _Float16 W1t[HID_ * OBSP_];     // W1 transposed [n][k], k padded w/ zeros
  _Float16 W2t[ADIMP_ * HID_];    // W2 transposed [n][k]
  _Float16 obs[MT_ * OBSP_];      // obs tile   [m][k] (f16)
  _Float16 hmat[MT_ * HID_];      // h tile     [m][k] (f16)
  float feats[4 * DCC_];          // surprise stats (shared across neurons)
  float sblk[C_ * DCC_];          // block-mean surprise scratch
  float xct[2][TILE_];            // double-buffered input tile (TDM target)
  float po[TILE_];                // previous mem_out (state; TDM store src)
  float xbuf[DM_];                // x = mean_c(xct)
  float rbuf[DM_];                // r
  float rpart[4 * DM_];           // partial sums for r
  float mi[DM_];
  float mo[DM_];
  float act[N_];
  float us[N_];
  float wv[N_];                   // softmax(w) scratch
  float scores[N_];               // scores / softmax(act) scratch
  float thmean[N_];               // mean(th, -1) per neuron (pre-relu)
  float escale[N_];               // exp(clip(temp,-4,4))
  float sigdec[N_];               // sigmoid(dec)
  float tempv[N_];
  float decv[N_];
  float red[N_];                  // reduction scratch
};

// ---------------- TDM helpers (CDNA5 ISA) ------
#if USE_TDM
typedef __attribute__((ext_vector_type(4))) unsigned int v4u;
typedef __attribute__((ext_vector_type(8))) int v8i_;
typedef __attribute__((ext_vector_type(4))) int v4i_;

// Build a D# for a 1-row, nelem x 4-byte contiguous tile.
__device__ inline void tdm_desc(v4u& g0, v8i_& g1, unsigned lds_addr,
                                unsigned long long ga, unsigned nelem) {
  g0.x = 1u;                                          // count=1 (valid), user D#
  g0.y = lds_addr;                                    // LDS byte address
  g0.z = (unsigned)(ga & 0xFFFFFFFFull);              // global_addr[31:0]
  g0.w = (unsigned)((ga >> 32) & 0x01FFFFFFull) | (2u << 30);  // [56:32]|type=2
  g1[0] = (int)(2u << 16);                            // wg_mask=0, data_size=4B
  g1[1] = (int)((nelem & 0xFFFFu) << 16);             // tensor_dim0[15:0]
  g1[2] = (int)(((nelem >> 16) & 0xFFFFu) | (1u << 16)); // dim0[31:16], dim1=1
  g1[3] = (int)((nelem & 0xFFFFu) << 16);             // tile_dim0 = nelem
  g1[4] = 1;                                          // tile_dim1=1, tile_dim2=0
  g1[5] = (int)nelem;                                 // dim0_stride[31:0]
  g1[6] = (int)((nelem & 0xFFFFu) << 16);             // stride hi=0 | dim1_stride lo
  g1[7] = (int)(nelem >> 16);                         // dim1_stride hi
}
__device__ inline unsigned lds_addr_of(const void* p) {
  // generic pointers into LDS carry the LDS byte offset in their low 32 bits
  return (unsigned)(unsigned long long)p;
}
__device__ inline void tdm_load(unsigned lds, const void* g, unsigned nelem) {
  v4u g0; v8i_ g1;
  v4i_ z4 = {0, 0, 0, 0};
  v8i_ z8 = {0, 0, 0, 0, 0, 0, 0, 0};
  tdm_desc(g0, g1, lds, (unsigned long long)g, nelem);
  __builtin_amdgcn_tensor_load_to_lds(g0, g1, z4, z4, z8, 0);
}
__device__ inline void tdm_store(unsigned lds, void* g, unsigned nelem) {
  v4u g0; v8i_ g1;
  v4i_ z4 = {0, 0, 0, 0};
  v8i_ z8 = {0, 0, 0, 0, 0, 0, 0, 0};
  tdm_desc(g0, g1, lds, (unsigned long long)g, nelem);
  __builtin_amdgcn_tensor_store_from_lds(g0, g1, z4, z4, z8, 0);
}
#endif

// ---------------- WMMA fragment loaders (CDNA5 ISA layouts) ------
// A: 16x32 f16, lane L<16 holds row M=L, K = k0+{0..7, 16..23};
//               lane L>=16 holds row M=L-16, K = k0+{8..15, 24..31}.
__device__ inline v16h load_a_frag(const _Float16* M, int ld, int m0, int k0, int lane) {
  int row = m0 + (lane & 15);
  int kb  = k0 + ((lane & 16) ? 8 : 0);
  const _Float16* p = M + row * ld + kb;
  v16h a;
#pragma unroll
  for (int e = 0; e < 8; ++e) { a[e] = p[e]; a[8 + e] = p[16 + e]; }
  return a;
}
// B: 32x16 f16 stored transposed in LDS as Wt[n][k]; lane L<16 -> col N=L,
// K = k0+{0..7, 16..23}; lane L>=16 -> col N=L-16, K = k0+{8..15, 24..31}.
__device__ inline v16h load_b_frag(const _Float16* Wt, int ld, int n0, int k0, int lane) {
  int row = n0 + (lane & 15);
  int kb  = k0 + ((lane & 16) ? 8 : 0);
  const _Float16* p = Wt + row * ld + kb;
  v16h b;
#pragma unroll
  for (int e = 0; e < 8; ++e) { b[e] = p[e]; b[8 + e] = p[16 + e]; }
  return b;
}

// ---------------- block reductions over 512-entry LDS arrays ----------------
__device__ inline float block_max_512(const float* a, float* red, int tid) {
  if (tid < 256) red[tid] = fmaxf(a[tid], a[tid + 256]);
  __syncthreads();
  if (tid < 32) {
    float v = red[tid];
#pragma unroll
    for (int j = 1; j < 8; ++j) v = fmaxf(v, red[tid + 32 * j]);
#pragma unroll
    for (int o = 16; o > 0; o >>= 1) v = fmaxf(v, __shfl_xor(v, o, 32));
    if (tid == 0) red[0] = v;
  }
  __syncthreads();
  float r = red[0];
  __syncthreads();
  return r;
}
__device__ inline float block_sum_512(const float* a, float* red, int tid) {
  if (tid < 256) red[tid] = a[tid] + a[tid + 256];
  __syncthreads();
  if (tid < 32) {
    float v = red[tid];
#pragma unroll
    for (int j = 1; j < 8; ++j) v += red[tid + 32 * j];
#pragma unroll
    for (int o = 16; o > 0; o >>= 1) v += __shfl_xor(v, o, 32);
    if (tid == 0) red[0] = v;
  }
  __syncthreads();
  float r = red[0];
  __syncthreads();
  return r;
}

// ---------------- persistent per-batch kernel -------------------------------
__global__ __launch_bounds__(NTHR_) void v8model_kernel(
    const float* __restrict__ cc,          // [BS,T,C,DM]
    const float* __restrict__ surprise,    // [BS,T,C,DCC]
    const float* __restrict__ primitives,  // [BS,N,DM]
    const float* __restrict__ thresholds,  // [BS,N,MC]
    const float* __restrict__ temperature, // [BS,N]
    const float* __restrict__ decay,       // [BS,N]
    const float* __restrict__ W1,          // [OBSD,HID]
    const float* __restrict__ b1,          // [HID]
    const float* __restrict__ W2,          // [HID,ADIM]
    const float* __restrict__ b2,          // [ADIM]
    const float* __restrict__ logstd,      // [ADIM]
    const unsigned char* __restrict__ eot, // [BS,T] bool
    float* __restrict__ out,               // [BS,T,C,DM] + [1] tlp
    float* __restrict__ Pws,               // [BS,N,DM] working copy
    float* __restrict__ thws)              // [BS,N,MC] working copy
{
  extern __shared__ char smem_raw[];
  Smem& S = *reinterpret_cast<Smem*>(smem_raw);
  const int tid  = threadIdx.x;
  const int wave = tid >> 5;
  const int lane = tid & 31;
  const int b    = blockIdx.x;

  float* Pb  = Pws  + (size_t)b * N_ * DM_;
  float* thb = thws + (size_t)b * N_ * MC_;

  // ---- one-time init: copy mutable params to workspace, weights to LDS f16
  {
    const float* Ps = primitives + (size_t)b * N_ * DM_;
    for (int i = tid; i < N_ * DM_; i += NTHR_) Pb[i] = Ps[i];
    const float* ts = thresholds + (size_t)b * N_ * MC_;
    for (int i = tid; i < N_ * MC_; i += NTHR_) thb[i] = ts[i];
    for (int i = tid; i < HID_ * OBSP_; i += NTHR_) {
      int n = i / OBSP_, k = i - n * OBSP_;
      S.W1t[i] = (_Float16)((k < OBSD_) ? W1[(size_t)k * HID_ + n] : 0.0f);
    }
    for (int i = tid; i < ADIMP_ * HID_; i += NTHR_) {
      int n = i >> 7, k = i & 127;
      S.W2t[i] = (_Float16)((n < ADIM_) ? W2[(size_t)k * ADIM_ + n] : 0.0f);
    }
    if (tid < N_) {
      S.act[tid] = 0.0f; S.us[tid] = 0.0f;
      S.tempv[tid] = temperature[(size_t)b * N_ + tid];
      S.decv[tid]  = decay[(size_t)b * N_ + tid];
    }
    if (tid < DM_) { S.mi[tid] = 0.0f; S.mo[tid] = 0.0f; }
    for (int i = tid; i < TILE_; i += NTHR_) S.po[i] = 0.0f;
    if (b == 0 && tid == 0) {
      float lp = 0.0f;
      for (int i = 0; i < ADIM_; ++i) lp += -logstd[i] - 0.5f * LOG2PI_F;
      out[(size_t)BS_ * T_ * C_ * DM_] = lp * (float)NB_;   // tlp
    }
  }
  __threadfence();
  __syncthreads();

#if USE_TDM
  // prime the input pipeline: DMA tile t=0 into buffer 0
  if (wave == 0)
    tdm_load(lds_addr_of(S.xct[0]), cc + (size_t)b * T_ * TILE_, TILE_);
#endif

  for (int blk = 0; blk < NB_; ++blk) {
    // ================= block start: surprise stats -> feats =================
    if (blk == 0) {
      if (tid < 4 * DCC_) S.feats[tid] = 0.0f;
    } else {
      if (tid < C_ * DCC_) {
        int c = tid >> 5, dcc = tid & 31;
        const float* sp = surprise +
            (((size_t)b * T_ + (size_t)(blk - 1) * AE_) * C_ + c) * DCC_ + dcc;
        float sum = 0.0f;
#pragma unroll 4
        for (int s2 = 0; s2 < AE_; ++s2) sum += sp[(size_t)s2 * C_ * DCC_];
        S.sblk[tid] = sum * (1.0f / AE_);
      }
      __syncthreads();
      if (tid < DCC_) {
        float mean = 0.0f, mx = -1e30f, mn = 1e30f;
#pragma unroll
        for (int c = 0; c < C_; ++c) {
          float v = S.sblk[c * DCC_ + tid];
          mean += v; mx = fmaxf(mx, v); mn = fminf(mn, v);
        }
        mean *= (1.0f / C_);
        float m2 = 0.0f;
#pragma unroll
        for (int c = 0; c < C_; ++c) {
          float dv = S.sblk[c * DCC_ + tid] - mean;
          m2 += dv * dv;
        }
        S.feats[tid]            = mean;
        S.feats[DCC_ + tid]     = mx;
        S.feats[2 * DCC_ + tid] = mn;
        S.feats[3 * DCC_ + tid] = sqrtf(m2 * (1.0f / C_));
      }
    }
    __syncthreads();

    // ================= MLP (WMMA) + parameter update, per 128-neuron tile ===
    for (int tile = 0; tile < N_ / MT_; ++tile) {
      // build obs tile in f16 (pre-update state; tiles are independent)
      for (int i = tid; i < MT_ * OBSP_; i += NTHR_) {
        int m = i / OBSP_, k = i - m * OBSP_;
        int n = tile * MT_ + m;
        float v;
        if (k < 4 * DCC_)      v = S.feats[k];
        else if (k == 128)     v = S.act[n];
        else if (k == 129)     v = S.tempv[n];
        else if (k == 130)     v = S.decv[n];
        else if (k == 131)     { float u = S.us[n]; v = u / (1.0f + u); }
        else                   v = 0.0f;
        S.obs[i] = (_Float16)v;
      }
      __syncthreads();

      // GEMM1: h = tanh(obs @ W1 + b1), 128x128 out, K=160 (5 WMMAs/subtile)
      for (int st = wave; st < (MT_ / 16) * (HID_ / 16); st += NTHR_ / 32) {
        int m0 = (st >> 3) << 4, n0 = (st & 7) << 4;
        v8f acc = {};
#pragma unroll
        for (int kk = 0; kk < OBSP_ / 32; ++kk) {
          v16h A = load_a_frag(S.obs, OBSP_, m0, kk * 32, lane);
          v16h B = load_b_frag(S.W1t, OBSP_, n0, kk * 32, lane);
          acc = __builtin_amdgcn_wmma_f32_16x16x32_f16(
              false, A, false, B, (short)0, acc, false, false);
        }
        int Nc = n0 + (lane & 15);
        int Mb = m0 + ((lane >> 4) << 3);
        float bb = b1[Nc];
#pragma unroll
        for (int v = 0; v < 8; ++v)
          S.hmat[(Mb + v) * HID_ + Nc] = (_Float16)fast_tanh(acc[v] + bb);
      }
      __syncthreads();

      // GEMM2: mean = h @ W2 + b2, 128x304 out, K=128 (4 WMMAs/subtile);
      // accumulator columns map straight onto the parameter updates.
      for (int st = wave; st < (MT_ / 16) * (ADIMP_ / 16); st += NTHR_ / 32) {
        int m0 = (st / 19) << 4, n0 = (st % 19) << 4;
        v8f acc = {};
#pragma unroll
        for (int kk = 0; kk < HID_ / 32; ++kk) {
          v16h A = load_a_frag(S.hmat, HID_, m0, kk * 32, lane);
          v16h B = load_b_frag(S.W2t, HID_, n0, kk * 32, lane);
          acc = __builtin_amdgcn_wmma_f32_16x16x32_f16(
              false, A, false, B, (short)0, acc, false, false);
        }
        int col = n0 + (lane & 15);
        int Mb  = m0 + ((lane >> 4) << 3);
        if (col < ADIM_) {
          float bb = b2[col];
#pragma unroll
          for (int v = 0; v < 8; ++v) {
            int n = tile * MT_ + Mb + v;
            float a = fminf(fmaxf(acc[v] + bb, -1.0f), 1.0f);
            if (col < DM_)            Pb[(size_t)n * DM_ + col] += a;
            else if (col < DM_ + MC_) thb[(size_t)n * MC_ + (col - DM_)] += a;
            else if (col == DM_ + MC_)     S.tempv[n] += a;
            else /* col == DM_+MC_+1 */    S.decv[n]  += a;
          }
        }
      }
      __syncthreads();
    }
    __threadfence();
    __syncthreads();

    // per-block derived quantities (post-update temp/dec/th)
    if (tid < N_) {
      const float* tr = thb + (size_t)tid * MC_;
      float s = 0.0f;
#pragma unroll
      for (int m = 0; m < MC_; ++m) s += tr[m];
      S.thmean[tid] = s * (1.0f / MC_);
      float tc = fminf(fmaxf(S.tempv[tid], -4.0f), 4.0f);
      S.escale[tid] = __expf(tc);
      S.sigdec[tid] = 1.0f / (1.0f + __expf(-S.decv[tid]));
    }
    __syncthreads();

    // ================= inner memory steps ==================================
    for (int s = 0; s < AE_; ++s) {
      int t = blk * AE_ + s;
      float keep = 1.0f;
      if (t > 0) keep = eot[(size_t)b * T_ + t - 1] ? 0.0f : 1.0f;

#if USE_TDM
      // wait for: output store of step t-1 (po is about to be overwritten)
      // and the DMA of this step's input tile; then kick off tile t+1.
      if (wave == 0) __builtin_amdgcn_s_wait_tensorcnt(0);
      __syncthreads();
      if (wave == 0 && t + 1 < T_)
        tdm_load(lds_addr_of(S.xct[(t + 1) & 1]),
                 cc + ((size_t)b * T_ + t + 1) * TILE_, TILE_);
      const float* xc = S.xct[t & 1];
#else
      const float* xsrc = cc + ((size_t)b * T_ + t) * TILE_;
      for (int i = tid; i < TILE_; i += NTHR_) S.xct[0][i] = xsrc[i];
      if (t + 1 < T_ && tid < 64)
        __builtin_prefetch(xsrc + (size_t)TILE_ + (size_t)tid * 32, 0, 1);
      const float* xc = S.xct[0];
#endif

      // stream reset
      if (tid < N_) { S.act[tid] *= keep; S.us[tid] *= keep; }
      if (tid < DM_) { S.mi[tid] *= keep; S.mo[tid] *= keep; }
      for (int i = tid; i < TILE_; i += NTHR_) S.po[i] *= keep;
      __syncthreads();

      // x = mean over C
      if (tid < DM_) {
        float sum = 0.0f;
#pragma unroll
        for (int c = 0; c < C_; ++c) sum += xc[c * DM_ + tid];
        S.xbuf[tid] = sum * (1.0f / C_);
      }
      __syncthreads();

      // sim = x . P[n] / 16 ; scores
      {
        int n = tid >> 1, hh = tid & 1;
        const float* Pr = Pb + (size_t)n * DM_ + hh * 128;
        const float* xr = S.xbuf + hh * 128;
        float partial = 0.0f;
#pragma unroll 4
        for (int d = 0; d < 128; ++d) partial = fmaf(Pr[d], xr[d], partial);
        partial += __shfl_xor(partial, 1, 32);
        if (hh == 0) {
          float sim = partial * 0.0625f;
          S.scores[n] = (sim - fmaxf(S.thmean[n], 0.0f)) * S.escale[n];
        }
      }
      __syncthreads();

      // w = softmax(scores); act, us updates
      float mx = block_max_512(S.scores, S.red, tid);
      if (tid < N_) S.wv[tid] = __expf(S.scores[tid] - mx);
      __syncthreads();
      float sm = block_sum_512(S.wv, S.red, tid);
      float inv = 1.0f / sm;
      if (tid < N_) {
        float w = S.wv[tid] * inv;
        S.wv[tid] = w;
        S.act[tid] = S.sigdec[tid] * S.act[tid] + w;
        S.us[tid] += w;
      }
      __syncthreads();

      // sm2 = softmax(act) (reuse S.scores)
      float mx2 = block_max_512(S.act, S.red, tid);
      if (tid < N_) S.scores[tid] = __expf(S.act[tid] - mx2);
      __syncthreads();
      float sm2 = block_sum_512(S.scores, S.red, tid);
      float inv2 = 1.0f / sm2;
      if (tid < N_) S.scores[tid] *= inv2;
      __syncthreads();

      // r = sm2 . P  (4 neuron chunks x 256 dims)
      {
        int d = tid & (DM_ - 1), ch = tid >> 8;
        const float* Pc = Pb + ((size_t)ch * 128) * DM_ + d;
        const float* sv = S.scores + ch * 128;
        float partial = 0.0f;
#pragma unroll 4
        for (int n2 = 0; n2 < 128; ++n2)
          partial = fmaf(sv[n2], Pc[(size_t)n2 * DM_], partial);
        S.rpart[ch * DM_ + d] = partial;
      }
      __syncthreads();
      if (tid < DM_) {
        float r = S.rpart[tid] + S.rpart[DM_ + tid] +
                  S.rpart[2 * DM_ + tid] + S.rpart[3 * DM_ + tid];
        S.rbuf[tid] = r;
        S.mi[tid] = 0.99f * S.mi[tid] + 0.01f * S.xbuf[tid];
        S.mo[tid] = 0.99f * S.mo[tid] + 0.01f * r;
      }
      __syncthreads();

      // mem_out = tanh(x_ct + r + 0.5*po); po <- mem_out; emit output tile
#if USE_TDM
      for (int i = tid; i < TILE_; i += NTHR_) {
        int d = i & (DM_ - 1);
        float v = fast_tanh(xc[i] + S.rbuf[d] + 0.5f * S.po[i]);
        S.po[i] = v;
      }
      __syncthreads();
      if (wave == 0)   // async DMA LDS -> global; waited at next step top
        tdm_store(lds_addr_of(S.po),
                  out + ((size_t)b * T_ + t) * TILE_, TILE_);
#else
      float* odst = out + ((size_t)b * T_ + t) * TILE_;
      for (int i = tid; i < TILE_; i += NTHR_) {
        int d = i & (DM_ - 1);
        float v = fast_tanh(xc[i] + S.rbuf[d] + 0.5f * S.po[i]);
        S.po[i] = v;
        odst[i] = v;
      }
      __syncthreads();
#endif
    }
  }
}

// ---------------- host-side launch ------------------------------------------
extern "C" void kernel_launch(void* const* d_in, const int* in_sizes, int n_in,
                              void* d_out, int out_size, void* d_ws, size_t ws_size,
                              hipStream_t stream) {
  (void)in_sizes; (void)n_in; (void)out_size; (void)ws_size;
  const float* cc          = (const float*)d_in[0];
  const float* surprise    = (const float*)d_in[1];
  const float* primitives  = (const float*)d_in[2];
  const float* thresholds  = (const float*)d_in[3];
  const float* temperature = (const float*)d_in[4];
  const float* decay       = (const float*)d_in[5];
  const float* W1          = (const float*)d_in[6];
  const float* b1          = (const float*)d_in[7];
  const float* W2          = (const float*)d_in[8];
  const float* b2          = (const float*)d_in[9];
  const float* logstd      = (const float*)d_in[10];
  const unsigned char* eot = (const unsigned char*)d_in[11];
  float* out = (float*)d_out;

  float* Pws  = (float*)d_ws;                       // BS*N*DM floats (4 MB)
  float* thws = Pws + (size_t)BS_ * N_ * DM_;       // BS*N*MC floats (0.5 MB)

  v8model_kernel<<<dim3(BS_), dim3(NTHR_), sizeof(Smem), stream>>>(
      cc, surprise, primitives, thresholds, temperature, decay,
      W1, b1, W2, b2, logstd, eot, out, Pws, thws);
}
